// DRAWModel_29961691857090
// MI455X (gfx1250) — matrix operand order repaired and checked by reading, not compiled, not run on hardware
//
#include <hip/hip_runtime.h>
#include <hip/hip_bf16.h>

typedef __bf16 bf16_t;
typedef __attribute__((ext_vector_type(16))) __bf16 v16bf;
typedef __attribute__((ext_vector_type(8)))  __bf16 v8bf;
typedef __attribute__((ext_vector_type(8)))  float   v8f;

#define NSTEPS 16
#define NB     512      // batch
#define IMG    4096     // 64*64
#define HID    512      // ENC == DEC
#define GD     2048     // 4*HID
#define ZD     128

__device__ __forceinline__ float sigm(float x) { return 1.f / (1.f + __expf(-x)); }

// ---------------- utility kernels ----------------
__global__ void k_cvt_bf16(const float* __restrict__ s, bf16_t* __restrict__ d, int n) {
  int i = blockIdx.x * blockDim.x + threadIdx.x;
  if (i < n) d[i] = (bf16_t)s[i];
}
__global__ void k_copy_f32(const float* __restrict__ s, float* __restrict__ d, int n) {
  int i = blockIdx.x * blockDim.x + threadIdx.x;
  if (i < n) d[i] = s[i];
}
__global__ void k_zero32(unsigned int* __restrict__ p, int n) {
  int i = blockIdx.x * blockDim.x + threadIdx.x;
  if (i < n) p[i] = 0u;
}

// ---------------- WMMA GEMM with register tile blocking ----------------
// C[M,N] = A1@W1^T (+ A2@W2^T) + bias. One wave per (MT*16)x(NT*16) C block.
// bf16 inputs, fp32 accumulate. Fragment layouts per CDNA5 ISA:
//   A (16x32 bf16): row = lane&15, khalf = lane>>4;
//     elems 0..7  -> k = kb + khalf*8 + e        (16 contiguous bytes)
//     elems 8..15 -> k = kb + 16 + khalf*8 + e   (16 contiguous bytes)
//   B = W^T (32x16): col = lane&15 (row of W); same k mapping as A
//   C/D (16x16 f32): elem i -> row = i + (lane>>4)*8, col = lane&15
template <int MT, int NT>
__device__ __forceinline__ void gemm_pass(v8f (&acc)[MT][NT],
                                          const bf16_t* __restrict__ A,
                                          const bf16_t* __restrict__ W,
                                          int K, int lda, int ldw,
                                          int tm, int tn, int half, int l15) {
  for (int kb = 0; kb < K; kb += 32) {
    union { v16bf v; v8bf h[2]; } a[MT], b[NT];
#pragma unroll
    for (int mi = 0; mi < MT; ++mi) {
      const bf16_t* ap = A + (size_t)((tm + mi) * 16 + l15) * lda + kb + half * 8;
      a[mi].h[0] = *(const v8bf*)ap;
      a[mi].h[1] = *(const v8bf*)(ap + 16);
    }
#pragma unroll
    for (int ni = 0; ni < NT; ++ni) {
      const bf16_t* wp = W + (size_t)((tn + ni) * 16 + l15) * ldw + kb + half * 8;
      b[ni].h[0] = *(const v8bf*)wp;
      b[ni].h[1] = *(const v8bf*)(wp + 16);
    }
#pragma unroll
    for (int mi = 0; mi < MT; ++mi)
#pragma unroll
      for (int ni = 0; ni < NT; ++ni)
        acc[mi][ni] = __builtin_amdgcn_wmma_f32_16x16x32_bf16(
            false, a[mi].v, false, b[ni].v, (short)0, acc[mi][ni], false, false);
  }
}

template <int MT, int NT>
__global__ void k_gemm(const bf16_t* __restrict__ A1, const bf16_t* __restrict__ W1,
                       int K1, int lda1, int ldw1,
                       const bf16_t* __restrict__ A2, const bf16_t* __restrict__ W2,
                       int K2, int lda2, int ldw2,
                       const float* __restrict__ bias,
                       float* __restrict__ C, int ldc) {
  const int lane = threadIdx.x & 31;
  const int half = lane >> 4;
  const int l15  = lane & 15;
  const int tm = blockIdx.y * MT;
  const int tn = blockIdx.x * NT;

  v8f acc[MT][NT];
#pragma unroll
  for (int ni = 0; ni < NT; ++ni) {
    const float bv = bias[(tn + ni) * 16 + l15];
#pragma unroll
    for (int mi = 0; mi < MT; ++mi)
#pragma unroll
      for (int i = 0; i < 8; ++i) acc[mi][ni][i] = bv;
  }

  gemm_pass<MT, NT>(acc, A1, W1, K1, lda1, ldw1, tm, tn, half, l15);
  if (A2) gemm_pass<MT, NT>(acc, A2, W2, K2, lda2, ldw2, tm, tn, half, l15);

#pragma unroll
  for (int mi = 0; mi < MT; ++mi)
#pragma unroll
    for (int ni = 0; ni < NT; ++ni) {
      float* cp = C + (size_t)((tm + mi) * 16 + half * 8) * ldc + (tn + ni) * 16 + l15;
#pragma unroll
      for (int i = 0; i < 8; ++i) cp[(size_t)i * ldc] = acc[mi][ni][i];
    }
}

// ---------------- LSTM elementwise ----------------
__global__ void k_lstm(const float* __restrict__ gates, float* __restrict__ cst,
                       float* __restrict__ h, bf16_t* __restrict__ h_bf,
                       bf16_t* __restrict__ extra_bf, int extra_ld, int extra_off) {
  int idx = blockIdx.x * blockDim.x + threadIdx.x;   // NB*HID
  int b = idx >> 9, j = idx & 511;
  const float* g = gates + (size_t)b * GD;
  float iv = sigm(g[j]);
  float fv = sigm(g[512 + j]);
  float gv = tanhf(g[1024 + j]);
  float ov = sigm(g[1536 + j]);
  float c2 = fv * cst[idx] + iv * gv;
  cst[idx] = c2;
  float hv = ov * tanhf(c2);
  h[idx] = hv;
  bf16_t hb = (bf16_t)hv;
  h_bf[idx] = hb;
  if (extra_bf) extra_bf[(size_t)b * extra_ld + extra_off + j] = hb;
}

// ---------------- z = mu + eps*exp(log_sigma) ----------------
__global__ void k_z(const float* __restrict__ pms, const float* __restrict__ eps_t,
                    bf16_t* __restrict__ z_bf) {
  int idx = blockIdx.x * blockDim.x + threadIdx.x;   // NB*ZD
  int b = idx >> 7, j = idx & 127;
  float mu = pms[(size_t)b * 256 + j];
  float ls = pms[(size_t)b * 256 + 128 + j];
  z_bf[idx] = (bf16_t)(mu + eps_t[idx] * __expf(ls));
}

// ---------------- attention read: filterbank + glimpse -> enc_in[:,0:128] ----------------
__global__ void k_attn_read(const float* __restrict__ x, const float* __restrict__ canvas,
                            const float* __restrict__ h_dec,
                            const float* __restrict__ attn_W, const float* __restrict__ attn_b,
                            bf16_t* __restrict__ enc_in) {
  __shared__ float sIm[4096];
  __shared__ float sXh[4096];
  __shared__ float sFy[512];
  __shared__ float sFx[512];
  __shared__ float sT1[512];
  __shared__ float sRed[256];
  __shared__ float sPar[5];
  __shared__ float sPrm[5];   // gx, gy, sigma2, delta, gamma
  const int b = blockIdx.x, tid = threadIdx.x;

  for (int i = tid; i < 4096; i += 256) {
    float xv = x[(size_t)b * IMG + i];
    sIm[i] = xv;
    sXh[i] = xv - sigm(canvas[(size_t)b * IMG + i]);
  }
  // p = h_dec @ attn_W^T + attn_b (5 dots over 512)
  float part[5] = {0.f, 0.f, 0.f, 0.f, 0.f};
  for (int h = tid; h < 512; h += 256) {
    float hv = h_dec[(size_t)b * HID + h];
#pragma unroll
    for (int k = 0; k < 5; ++k) part[k] += hv * attn_W[k * 512 + h];
  }
  for (int k = 0; k < 5; ++k) {
    sRed[tid] = part[k];
    __syncthreads();
    for (int s = 128; s > 0; s >>= 1) {
      if (tid < s) sRed[tid] += sRed[tid + s];
      __syncthreads();
    }
    if (tid == 0) sPar[k] = sRed[0] + attn_b[k];
    __syncthreads();
  }
  if (tid == 0) {
    sPrm[0] = 65.f * 0.5f * (sPar[0] + 1.f);          // gx
    sPrm[1] = 65.f * 0.5f * (sPar[1] + 1.f);          // gy
    sPrm[2] = __expf(sPar[2]);                        // sigma2
    sPrm[3] = 63.f * __expf(sPar[3]) / 7.f;           // delta (N=8)
    sPrm[4] = __expf(sPar[4]);                        // gamma
  }
  __syncthreads();
  const float gx = sPrm[0], gy = sPrm[1], s2 = sPrm[2], delta = sPrm[3], gamma = sPrm[4];
  // filterbanks (N=8, 64 pixels)
  for (int i = tid; i < 512; i += 256) {
    int n = i >> 6, a = i & 63;
    float mux = gx + ((float)n - 4.5f) * delta;
    float muy = gy + ((float)n - 4.5f) * delta;
    float dx = (float)a - mux, dy = (float)a - muy;
    sFx[i] = __expf(-dx * dx / (2.f * s2));
    sFy[i] = __expf(-dy * dy / (2.f * s2));
  }
  __syncthreads();
  if (tid < 16) {
    float* base = (tid < 8) ? (sFx + tid * 64) : (sFy + (tid - 8) * 64);
    float s = 0.f;
    for (int j = 0; j < 64; ++j) s += base[j];
    s = 1.f / (s + 1e-8f);
    for (int j = 0; j < 64; ++j) base[j] *= s;
  }
  __syncthreads();
  // glimpse: g = Fy @ im @ Fx^T, for img then x_hat
  for (int pass = 0; pass < 2; ++pass) {
    const float* im = pass ? sXh : sIm;
    for (int i = tid; i < 512; i += 256) {     // t1[n][w] = sum_h Fy[n,h]*im[h,w]
      int n = i >> 6, w = i & 63;
      float s = 0.f;
      for (int h = 0; h < 64; ++h) s += sFy[n * 64 + h] * im[h * 64 + w];
      sT1[i] = s;
    }
    __syncthreads();
    if (tid < 64) {                            // g[n][m] = sum_w t1[n,w]*Fx[m,w]
      int n = tid >> 3, m = tid & 7;
      float s = 0.f;
      for (int w = 0; w < 64; ++w) s += sT1[n * 64 + w] * sFx[m * 64 + w];
      enc_in[(size_t)b * 640 + pass * 64 + tid] = (bf16_t)(gamma * s);
    }
    __syncthreads();
  }
}

// ---------------- attention write: canvas += Fy2^T @ w @ Fx2 / gamma; emit d_out[t] ----------------
__global__ void k_attn_write(const float* __restrict__ h_dec,
                             const float* __restrict__ attn_W, const float* __restrict__ attn_b,
                             const float* __restrict__ wbuf, float* __restrict__ canvas,
                             float* __restrict__ out_t) {
  __shared__ float sFy[512];
  __shared__ float sFx[512];
  __shared__ float sW[64];
  __shared__ float sT2[512];
  __shared__ float sRed[256];
  __shared__ float sPar[5];
  __shared__ float sPrm[5];
  const int b = blockIdx.x, tid = threadIdx.x;

  if (tid < 64) sW[tid] = wbuf[(size_t)b * 64 + tid];
  float part[5] = {0.f, 0.f, 0.f, 0.f, 0.f};
  for (int h = tid; h < 512; h += 256) {
    float hv = h_dec[(size_t)b * HID + h];
#pragma unroll
    for (int k = 0; k < 5; ++k) part[k] += hv * attn_W[k * 512 + h];
  }
  for (int k = 0; k < 5; ++k) {
    sRed[tid] = part[k];
    __syncthreads();
    for (int s = 128; s > 0; s >>= 1) {
      if (tid < s) sRed[tid] += sRed[tid + s];
      __syncthreads();
    }
    if (tid == 0) sPar[k] = sRed[0] + attn_b[k];
    __syncthreads();
  }
  if (tid == 0) {
    sPrm[0] = 65.f * 0.5f * (sPar[0] + 1.f);
    sPrm[1] = 65.f * 0.5f * (sPar[1] + 1.f);
    sPrm[2] = __expf(sPar[2]);
    sPrm[3] = 63.f * __expf(sPar[3]) / 7.f;
    sPrm[4] = __expf(sPar[4]);
  }
  __syncthreads();
  const float gx = sPrm[0], gy = sPrm[1], s2 = sPrm[2], delta = sPrm[3];
  const float inv_g = 1.f / sPrm[4];
  for (int i = tid; i < 512; i += 256) {
    int n = i >> 6, a = i & 63;
    float mux = gx + ((float)n - 4.5f) * delta;
    float muy = gy + ((float)n - 4.5f) * delta;
    float dx = (float)a - mux, dy = (float)a - muy;
    sFx[i] = __expf(-dx * dx / (2.f * s2));
    sFy[i] = __expf(-dy * dy / (2.f * s2));
  }
  __syncthreads();
  if (tid < 16) {
    float* base = (tid < 8) ? (sFx + tid * 64) : (sFy + (tid - 8) * 64);
    float s = 0.f;
    for (int j = 0; j < 64; ++j) s += base[j];
    s = 1.f / (s + 1e-8f);
    for (int j = 0; j < 64; ++j) base[j] *= s;
  }
  __syncthreads();
  // t2[h][m] = sum_n Fy[n,h] * w[n,m]
  for (int i = tid; i < 512; i += 256) {
    int h = i >> 3, m = i & 7;
    float s = 0.f;
#pragma unroll
    for (int n = 0; n < 8; ++n) s += sFy[n * 64 + h] * sW[n * 8 + m];
    sT2[i] = s;
  }
  __syncthreads();
  // out[h][a] = sum_m t2[h,m] * Fx[m,a]; canvas update + emit
  for (int i = tid; i < 4096; i += 256) {
    int h = i >> 6, a = i & 63;
    float s = 0.f;
#pragma unroll
    for (int m = 0; m < 8; ++m) s += sT2[h * 8 + m] * sFx[m * 64 + a];
    float nc = canvas[(size_t)b * IMG + i] + s * inv_g;
    canvas[(size_t)b * IMG + i] = nc;
    out_t[(size_t)b * IMG + i] = nc;
  }
}

extern "C" void kernel_launch(void* const* d_in, const int* in_sizes, int n_in,
                              void* d_out, int out_size, void* d_ws, size_t ws_size,
                              hipStream_t stream) {
  (void)in_sizes; (void)n_in; (void)out_size; (void)ws_size;
  const float* x       = (const float*)d_in[0];
  const float* eps     = (const float*)d_in[1];
  const float* enc_Wih = (const float*)d_in[2];
  const float* enc_Whh = (const float*)d_in[3];
  const float* enc_b   = (const float*)d_in[4];
  const float* mu_W    = (const float*)d_in[5];
  const float* mu_b    = (const float*)d_in[6];
  const float* sig_W   = (const float*)d_in[7];
  const float* sig_b   = (const float*)d_in[8];
  const float* dec_Wih = (const float*)d_in[9];
  const float* dec_Whh = (const float*)d_in[10];
  const float* dec_b   = (const float*)d_in[11];
  const float* write_W = (const float*)d_in[12];
  const float* write_b = (const float*)d_in[13];
  const float* attn_W  = (const float*)d_in[14];
  const float* attn_b  = (const float*)d_in[15];
  float* out = (float*)d_out;

  char* ws = (char*)d_ws;
  size_t off = 0;
  auto alloc = [&](size_t bytes) -> char* {
    char* p = ws + off;
    off += (bytes + 255) & ~(size_t)255;
    return p;
  };
  // bf16 weights
  bf16_t* wEncIh = (bf16_t*)alloc((size_t)GD * 640 * 2);
  bf16_t* wEncHh = (bf16_t*)alloc((size_t)GD * 512 * 2);
  bf16_t* wDecIh = (bf16_t*)alloc((size_t)GD * 128 * 2);
  bf16_t* wDecHh = (bf16_t*)alloc((size_t)GD * 512 * 2);
  bf16_t* wMuSig = (bf16_t*)alloc((size_t)256 * 512 * 2);
  bf16_t* wWrite = (bf16_t*)alloc((size_t)64 * 512 * 2);
  float*  bMuSig = (float*)alloc(256 * 4);
  // fp32 state (contiguous, zeroed once per launch)
  float* hEnc   = (float*)alloc((size_t)NB * HID * 4);
  float* cEnc   = (float*)alloc((size_t)NB * HID * 4);
  float* hDec   = (float*)alloc((size_t)NB * HID * 4);
  float* cDec   = (float*)alloc((size_t)NB * HID * 4);
  float* canvas = (float*)alloc((size_t)NB * IMG * 4);
  // bf16 state (contiguous, zeroed)
  bf16_t* encIn = (bf16_t*)alloc((size_t)NB * 640 * 2);
  bf16_t* hEncB = (bf16_t*)alloc((size_t)NB * HID * 2);
  bf16_t* hDecB = (bf16_t*)alloc((size_t)NB * HID * 2);
  bf16_t* zB    = (bf16_t*)alloc((size_t)NB * ZD * 2);
  // scratch
  float* gates = (float*)alloc((size_t)NB * GD * 4);
  float* pms   = (float*)alloc((size_t)NB * 256 * 4);
  float* wbuf  = (float*)alloc((size_t)NB * 64 * 4);

  const int TB = 256;
  auto cdiv = [](int a, int b) { return (a + b - 1) / b; };

  // --- pre-pass: weight conversion + state zero ---
  k_cvt_bf16<<<cdiv(GD * 640, TB), TB, 0, stream>>>(enc_Wih, wEncIh, GD * 640);
  k_cvt_bf16<<<cdiv(GD * 512, TB), TB, 0, stream>>>(enc_Whh, wEncHh, GD * 512);
  k_cvt_bf16<<<cdiv(GD * 128, TB), TB, 0, stream>>>(dec_Wih, wDecIh, GD * 128);
  k_cvt_bf16<<<cdiv(GD * 512, TB), TB, 0, stream>>>(dec_Whh, wDecHh, GD * 512);
  k_cvt_bf16<<<cdiv(128 * 512, TB), TB, 0, stream>>>(mu_W, wMuSig, 128 * 512);
  k_cvt_bf16<<<cdiv(128 * 512, TB), TB, 0, stream>>>(sig_W, wMuSig + 128 * 512, 128 * 512);
  k_cvt_bf16<<<cdiv(64 * 512, TB), TB, 0, stream>>>(write_W, wWrite, 64 * 512);
  k_copy_f32<<<1, 128, 0, stream>>>(mu_b, bMuSig, 128);
  k_copy_f32<<<1, 128, 0, stream>>>(sig_b, bMuSig + 128, 128);
  const int zf = (4 * NB * HID * 4 + NB * IMG * 4) / 4;               // fp32 state words
  const int zb = (NB * 640 * 2 + 2 * NB * HID * 2 + NB * ZD * 2) / 4; // bf16 state words
  k_zero32<<<cdiv(zf, TB), TB, 0, stream>>>((unsigned int*)hEnc, zf);
  k_zero32<<<cdiv(zb, TB), TB, 0, stream>>>((unsigned int*)encIn, zb);

  // --- T=16 recurrent steps ---
  for (int t = 0; t < NSTEPS; ++t) {
    k_attn_read<<<NB, 256, 0, stream>>>(x, canvas, hDec, attn_W, attn_b, encIn);
    // encoder gates: [NB,640]@Wih^T + [NB,512]@Whh^T + b  (2x4 register tiles)
    k_gemm<2, 4><<<dim3(GD / 64, NB / 32), 32, 0, stream>>>(
        encIn, wEncIh, 640, 640, 640, hEncB, wEncHh, 512, 512, 512, enc_b, gates, GD);
    k_lstm<<<NB * HID / TB, TB, 0, stream>>>(gates, cEnc, hEnc, hEncB,
                                             (bf16_t*)nullptr, 0, 0);
    // mu | log_sigma: [NB,512]@[256,512]^T  (2x2 tiles)
    k_gemm<2, 2><<<dim3(256 / 32, NB / 32), 32, 0, stream>>>(
        hEncB, wMuSig, 512, 512, 512, (const bf16_t*)nullptr, (const bf16_t*)nullptr,
        0, 0, 0, bMuSig, pms, 256);
    k_z<<<cdiv(NB * ZD, TB), TB, 0, stream>>>(pms, eps + (size_t)t * NB * ZD, zB);
    // decoder gates: [NB,128]@Wih^T + [NB,512]@Whh^T + b  (2x4 tiles)
    k_gemm<2, 4><<<dim3(GD / 64, NB / 32), 32, 0, stream>>>(
        zB, wDecIh, 128, 128, 128, hDecB, wDecHh, 512, 512, 512, dec_b, gates, GD);
    k_lstm<<<NB * HID / TB, TB, 0, stream>>>(gates, cDec, hDec, hDecB, encIn, 640, 128);
    // write patch: [NB,512]@[64,512]^T  (1x1 tiles keeps grid populated)
    k_gemm<1, 1><<<dim3(64 / 16, NB / 16), 32, 0, stream>>>(
        hDecB, wWrite, 512, 512, 512, (const bf16_t*)nullptr, (const bf16_t*)nullptr,
        0, 0, 0, write_b, wbuf, 64);
    k_attn_write<<<NB, 256, 0, stream>>>(hDec, attn_W, attn_b, wbuf, canvas,
                                         out + (size_t)t * NB * IMG);
  }
}
